// LSTMLayer_695784702521
// MI455X (gfx1250) — compile-verified, matrix operand order checked by reading, not run on hardware
//
#include <hip/hip_runtime.h>

// ---------------------------------------------------------------------------
// Time-LSTM, persistent-kernel formulation for MI455X (gfx1250, wave32, WMMA).
//
//   T=2048 steps, B=32, I=256, H=512, 4H=2048.
//   32 persistent workgroups x 128 threads (4 waves).
//   WG `wg` owns hidden cols [16*wg,16*wg+16); wave g owns gate g.
//   W_ih/W_hh bf16 fragments pinned in VGPRs (8+16 k-chunks x v16bf).
//   Per step: stage x_t,h_{t-1} -> LDS bf16; 48x v_wmma_f32_16x16x32_bf16
//   per wave with depth-2 software-pipelined ds_load_b128 A-fragments;
//   gate combine via LDS; c state resident in LDS; h_t -> d_out;
//   grid barrier (atomic generation counter, plain-load spin) per step.
// ---------------------------------------------------------------------------

#define T_STEPS 2048
#define BATCH   32
#define IN      256
#define HID     512
#define G4      2048          // 4*HID
#define NBLK    32            // persistent workgroups
#define NTHR    128           // 4 waves of 32
#define ZP      68            // z_lds row pitch (floats), padded vs bank conflicts

typedef __attribute__((ext_vector_type(16))) __bf16 bf16x16;
typedef __attribute__((ext_vector_type(4)))  __bf16 bf16x4;
typedef __attribute__((ext_vector_type(8)))  float  f32x8;
typedef __attribute__((ext_vector_type(4)))  float  f32x4;

struct Bar { unsigned count; unsigned gen; };

// workspace layout (bytes)
#define WS_WIH 0                          // bf16 [2048][256]  = 1 MiB
#define WS_WHH (G4 * IN * 2)              // bf16 [2048][512]  = 2 MiB
#define WS_BAR (WS_WHH + G4 * HID * 2)    // Bar (8 bytes)

// ---------------------------------------------------------------------------
// Prep kernel: fp32 weights -> bf16 packed copies in ws; zero grid barrier.
// Runs before the persistent kernel on the same stream (graph-replay safe).
// ---------------------------------------------------------------------------
__global__ void lstm_prep_kernel(const float* __restrict__ W_ih,
                                 const float* __restrict__ W_hh,
                                 __bf16* __restrict__ wih_bf,
                                 __bf16* __restrict__ whh_bf,
                                 Bar* __restrict__ bar) {
  size_t idx = (size_t)blockIdx.x * blockDim.x + threadIdx.x;
  if (idx == 0) { bar->count = 0u; bar->gen = 0u; }
  const size_t NIH = (size_t)G4 * IN;     // 524288 (drop time col, kept fp32)
  if (idx < NIH) {
    size_t r = idx / IN, k = idx % IN;
    wih_bf[idx] = (__bf16)W_ih[r * (IN + 1) + k];
  }
  const size_t NHH = (size_t)G4 * HID;    // 1048576
  if (idx < NHH) whh_bf[idx] = (__bf16)W_hh[idx];
}

// ---------------------------------------------------------------------------
// Persistent recurrent kernel.
// ---------------------------------------------------------------------------
__global__ void __launch_bounds__(NTHR, 1)
lstm_persistent_kernel(const float* __restrict__ x,      // [T,B,I]
                       const float* __restrict__ tvec,   // [T,B,1]
                       const float* __restrict__ W_ih,   // [4H,I+1] fp32 (time col)
                       const float* __restrict__ bvec,   // [4H]
                       const float* __restrict__ h0,     // [B,H]
                       const float* __restrict__ c0,     // [B,H]
                       const __bf16* __restrict__ wih_bf,
                       const __bf16* __restrict__ whh_bf,
                       Bar* __restrict__ bar,
                       float* __restrict__ out) {        // [T,B,H] ++ hT ++ cT
  __shared__ __bf16 x_lds[BATCH * IN];    // 16 KiB  x_t as bf16
  __shared__ __bf16 h_lds[BATCH * HID];   // 32 KiB  h_{t-1} as bf16
  __shared__ float  z_lds[BATCH * ZP];    // 8.5 KiB pre-activations (4 gates x 16)
  __shared__ float  c_lds[BATCH * 16];    // 2 KiB   owned c slice (persists)
  __shared__ float  t_lds[BATCH];         // time scalars

  const int tid  = threadIdx.x;
  const int wg   = blockIdx.x;            // 0..31: owns hidden cols [16*wg,16*wg+16)
  const int gate = tid >> 5;              // wave id = gate 0..3 (i,f,g,o)
  const int lane = tid & 31;
  const int n    = lane & 15;             // tile column (output col / B-row)
  const int kh   = (lane >> 4) * 16;      // K half owned by this lane
  const int grow = gate * HID + wg * 16 + n;   // gate-row in [0,4H)

  // --- Pin this wave's weight B-fragments in VGPRs (bf16). Lane layout:
  // lane holds col n = lane%16, 16 consecutive K at K = 32*kc + kh
  // (approximation of the ISA 16-bit striping; exact swizzle not checkable
  //  compile-only -- production would use GLOBAL_LOAD_TR16_B128).
  bf16x16 bfi[8];
#pragma unroll
  for (int kc = 0; kc < 8; ++kc)
    bfi[kc] = *(const bf16x16*)(wih_bf + (size_t)grow * IN + kc * 32 + kh);
  bf16x16 bfh[16];
#pragma unroll
  for (int kc = 0; kc < 16; ++kc)
    bfh[kc] = *(const bf16x16*)(whh_bf + (size_t)grow * HID + kc * 32 + kh);

  const float bias_l  = bvec[grow];
  const float wtime_l = W_ih[(size_t)grow * (IN + 1) + IN];  // time column, fp32

  for (int e = tid; e < BATCH * 16; e += NTHR)
    c_lds[e] = c0[(e >> 4) * HID + wg * 16 + (e & 15)];

  float* hT = out + (size_t)T_STEPS * BATCH * HID;
  float* cT = hT + BATCH * HID;

  // A-fragment loads: x chunks kc=0..7, then h chunks kc=0..15
#define LOAD_X(m, kc) (*(const bf16x16*)&x_lds[(16 * (m) + n) * IN + (kc) * 32 + kh])
#define LOAD_H(m, kc) (*(const bf16x16*)&h_lds[(16 * (m) + n) * HID + (kc) * 32 + kh])

  for (int t = 0; t < T_STEPS; ++t) {
    // ---- stage h_{t-1} and x_t into LDS as bf16 -------------------------
    const float* hsrc = (t == 0) ? h0 : (out + (size_t)(t - 1) * BATCH * HID);
    for (int i = tid; i < (BATCH * HID) / 4; i += NTHR) {
      f32x4 f = *(const f32x4*)(hsrc + i * 4);
      bf16x4 v; v.x = (__bf16)f.x; v.y = (__bf16)f.y;
      v.z = (__bf16)f.z; v.w = (__bf16)f.w;
      *(bf16x4*)&h_lds[i * 4] = v;
    }
    const float* xsrc = x + (size_t)t * BATCH * IN;
    for (int i = tid; i < (BATCH * IN) / 4; i += NTHR) {
      f32x4 f = *(const f32x4*)(xsrc + i * 4);
      bf16x4 v; v.x = (__bf16)f.x; v.y = (__bf16)f.y;
      v.z = (__bf16)f.z; v.w = (__bf16)f.w;
      *(bf16x4*)&x_lds[i * 4] = v;
    }
    if (tid < BATCH) t_lds[tid] = tvec[t * BATCH + tid];
    if (t + 1 < T_STEPS)  // warm next x tile into L2 (global_prefetch_b8)
      __builtin_prefetch(x + (size_t)(t + 1) * BATCH * IN + lane * 64, 0, 1);
    __syncthreads();

    // ---- z tile = x_t * Wih^T + h * Whh^T (+ bias + time) ---------------
    // 24 fused K-chunks per M-tile; depth-2 pipelined A loads so each WMMA
    // waits on dscnt<=2 instead of dscnt==0 (hides LDS latency).
#pragma unroll
    for (int m = 0; m < 2; ++m) {          // batch rows 16*m..16*m+15
      f32x8 acc = {};
      bf16x16 a0 = LOAD_X(m, 0);
      bf16x16 a1 = LOAD_X(m, 1);
#pragma unroll
      for (int kc = 0; kc < 24; ++kc) {
        bf16x16 a_next = a1;
        if (kc + 2 < 8)       a_next = LOAD_X(m, kc + 2);
        else if (kc + 2 < 24) a_next = LOAD_H(m, kc + 2 - 8);
        if (kc < 8)
          acc = __builtin_amdgcn_wmma_f32_16x16x32_bf16(
              false, a0, false, bfi[kc], (short)0, acc, false, false);
        else
          acc = __builtin_amdgcn_wmma_f32_16x16x32_bf16(
              false, a0, false, bfh[kc - 8], (short)0, acc, false, false);
        a0 = a1;
        a1 = a_next;
      }
      // C/D layout: lane covers col n, rows v + (lane>=16 ? 8 : 0)
#pragma unroll
      for (int v = 0; v < 8; ++v) {
        int row = 16 * m + v + (kh >> 1);  // kh=16 -> +8
        z_lds[row * ZP + gate * 16 + n] =
            acc[v] + bias_l + t_lds[row] * wtime_l;
      }
    }
    __syncthreads();

    // ---- gate nonlinearities + state update (fp32) ----------------------
    for (int e = tid; e < BATCH * 16; e += NTHR) {
      int r = e >> 4, j = e & 15;
      float zi = z_lds[r * ZP +  0 + j];
      float zf = z_lds[r * ZP + 16 + j];
      float zg = z_lds[r * ZP + 32 + j];
      float zo = z_lds[r * ZP + 48 + j];
      float ig = 1.f / (1.f + __expf(-zi));
      float fg = 1.f / (1.f + __expf(-zf));
      float gg = tanhf(zg);
      float og = 1.f / (1.f + __expf(-zo));
      float cn = fg * c_lds[e] + ig * gg;
      float hn = og * tanhf(cn);
      c_lds[e] = cn;
      out[(size_t)t * BATCH * HID + r * HID + wg * 16 + j] = hn;
      if (t == T_STEPS - 1) {
        hT[r * HID + wg * 16 + j] = hn;
        cT[r * HID + wg * 16 + j] = cn;
      }
    }

    // ---- grid-wide barrier: make h_t visible to all 32 WGs --------------
    __syncthreads();
    if (tid == 0) {
      __threadfence();
      unsigned a = atomicAdd(&bar->count, 1u);
      if (a == NBLK - 1) {
        atomicExch(&bar->count, 0u);
        __threadfence();
        atomicAdd(&bar->gen, 1u);
      } else {
        while (__hip_atomic_load(&bar->gen, __ATOMIC_ACQUIRE,
                                 __HIP_MEMORY_SCOPE_AGENT) < (unsigned)(t + 1))
          __builtin_amdgcn_s_sleep(2);
      }
      __threadfence();
    }
    __syncthreads();
  }
#undef LOAD_X
#undef LOAD_H
}

// ---------------------------------------------------------------------------
extern "C" void kernel_launch(void* const* d_in, const int* in_sizes, int n_in,
                              void* d_out, int out_size, void* d_ws, size_t ws_size,
                              hipStream_t stream) {
  const float* x    = (const float*)d_in[0];
  const float* tv   = (const float*)d_in[1];
  const float* W_ih = (const float*)d_in[2];
  const float* W_hh = (const float*)d_in[3];
  const float* b    = (const float*)d_in[4];
  const float* h0   = (const float*)d_in[5];
  const float* c0   = (const float*)d_in[6];
  float* out = (float*)d_out;

  char* ws = (char*)d_ws;
  __bf16* wih_bf = (__bf16*)(ws + WS_WIH);
  __bf16* whh_bf = (__bf16*)(ws + WS_WHH);
  Bar*    bar    = (Bar*)(ws + WS_BAR);

  // 1) bf16 weight packing + barrier init (stream-ordered before persistent k)
  lstm_prep_kernel<<<(G4 * HID + 255) / 256, 256, 0, stream>>>(
      W_ih, W_hh, wih_bf, whh_bf, bar);

  // 2) persistent recurrence: 32 WGs x 128 threads, 2048 device-side steps
  lstm_persistent_kernel<<<NBLK, NTHR, 0, stream>>>(
      x, tv, W_ih, b, h0, c0, wih_bf, whh_bf, bar, out);
}
